// Attention_45878840656175
// MI455X (gfx1250) — matrix operand order repaired and checked by reading, not compiled, not run on hardware
//
#include <hip/hip_runtime.h>

typedef __attribute__((ext_vector_type(16))) __bf16 v16bf;
typedef __attribute__((ext_vector_type(8)))  float  v8f;
typedef unsigned int   u32;
typedef unsigned short u16;
typedef unsigned long long u64;

// ---------- helpers ----------
__device__ __forceinline__ u16 f2bf(float f) {
    u32 u = __float_as_uint(f);
    u32 r = u + 0x7FFFu + ((u >> 16) & 1u);   // round-to-nearest-even
    return (u16)(r >> 16);
}
__device__ __forceinline__ u32 pack2(u16 lo, u16 hi) {
    return (u32)lo | ((u32)hi << 16);
}
// LDS byte offset of a __shared__ pointer (flat addr low 32 bits = LDS offset)
__device__ __forceinline__ u32 lds_off(const void* p) {
    return (u32)(uintptr_t)p;
}
// async copy 16B global -> LDS (tracked by ASYNCcnt); lds must be 16B aligned
__device__ __forceinline__ void async_copy_b128(u32 lds, const void* g) {
    asm volatile("global_load_async_to_lds_b128 %0, %1, off"
                 :: "v"(lds), "v"((u64)(uintptr_t)g) : "memory");
}
__device__ __forceinline__ void wait_async0() {
    asm volatile("s_wait_asynccnt 0" ::: "memory");
}

union Frag { v16bf v; u32 u[8]; };

// A-operand (16x32 bf16): lane holds row M=lane&15; sel=lane>>4 picks K subsets
__device__ __forceinline__ v16bf ldfragA(const u32* row, int sel) {
    Frag f;
#pragma unroll
    for (int v = 0; v < 4; ++v) f.u[v]     = row[sel * 4 + v];
#pragma unroll
    for (int v = 0; v < 4; ++v) f.u[4 + v] = row[8 + sel * 4 + v];
    return f.v;
}
// B-operand (32x16 bf16): lane holds col N=lane&15; sel picks K subsets
__device__ __forceinline__ v16bf ldfragB(const u32* row, int sel) {
    Frag f;
#pragma unroll
    for (int v = 0; v < 8; ++v) f.u[v] = row[(v >> 2) * 8 + sel * 4 + (v & 3)];
    return f.v;
}

__device__ __forceinline__ v8f wmma_bf16(v16bf a, v16bf b, v8f c) {
    return __builtin_amdgcn_wmma_f32_16x16x32_bf16(
        false, a, false, b, (short)0, c, false, false);
}

// ---------- GEMM: C(bf16) = A(f32) @ B(f32) + bias, tiles 128x128x32 ----------
#define GBM 128
#define GBN 128
#define GBK 32
#define GPAD 17   // 16 k-pair uints + 1 pad (VALU-filled tiles)

__global__ __launch_bounds__(256) void qkv_gemm_kernel(
    const float* __restrict__ X, const float* __restrict__ W,
    const float* __restrict__ bias, u16* __restrict__ Out,
    int M, int N, int K)
{
    __shared__ u32 sA[GBM][GPAD];   // [m][k-pair]
    __shared__ u32 sB[GBN][GPAD];   // [n][k-pair] (B transposed into LDS)

    const int tid  = threadIdx.x;
    const int wave = tid >> 5, lane = tid & 31;
    const int sel  = lane >> 4,  l15 = lane & 15;
    const int wm = wave >> 2, wn = wave & 3;       // 2x4 wave grid
    const int bM = blockIdx.y * GBM, bN = blockIdx.x * GBN;

    v8f acc[4][2];
#pragma unroll
    for (int i = 0; i < 4; ++i)
#pragma unroll
        for (int j = 0; j < 2; ++j)
#pragma unroll
            for (int r = 0; r < 8; ++r) acc[i][j][r] = 0.0f;

    const int arow = tid >> 1, ahalf = tid & 1;          // A loader
    const int bnb  = (tid & 15) * 8, bkp = tid >> 4;     // B loader (k-pair)

    for (int k0 = 0; k0 < K; k0 += GBK) {
        __syncthreads();
        const float* ap = X + (size_t)(bM + arow) * K + k0 + ahalf * 16;
#pragma unroll
        for (int i = 0; i < 8; ++i)
            sA[arow][ahalf * 8 + i] = pack2(f2bf(ap[2 * i]), f2bf(ap[2 * i + 1]));
        const float* bp0 = W + (size_t)(k0 + 2 * bkp) * N + bN + bnb;
        const float* bp1 = bp0 + N;
#pragma unroll
        for (int i = 0; i < 8; ++i)
            sB[bnb + i][bkp] = pack2(f2bf(bp0[i]), f2bf(bp1[i]));
        if (k0 + GBK < K) {                       // warm L2 one k-step ahead
            __builtin_prefetch(ap + GBK, 0, 1);
            __builtin_prefetch(bp0 + (size_t)GBK * N, 0, 1);
        }
        __syncthreads();

        v16bf af[4];
#pragma unroll
        for (int i = 0; i < 4; ++i)
            af[i] = ldfragA(&sA[wm * 64 + i * 16 + l15][0], sel);
#pragma unroll
        for (int j = 0; j < 2; ++j) {
            v16bf bf = ldfragB(&sB[wn * 32 + j * 16 + l15][0], sel);
#pragma unroll
            for (int i = 0; i < 4; ++i)
                acc[i][j] = wmma_bf16(af[i], bf, acc[i][j]);
        }
    }

#pragma unroll
    for (int i = 0; i < 4; ++i)
#pragma unroll
        for (int j = 0; j < 2; ++j) {
            int gc = bN + wn * 32 + j * 16 + l15;
            float bv = bias[gc];
#pragma unroll
            for (int r = 0; r < 8; ++r) {
                int gr = bM + wm * 64 + i * 16 + r + sel * 8;
                Out[(size_t)gr * N + gc] = f2bf(acc[i][j][r] + bv);
            }
        }
}

// ---------- GEMM: C(f32) = A(bf16) @ B(f32) + bias  (output projection) ----------
#define OAPAD 20  // 80B row stride: 16B-aligned for async B128, conflict-free

__global__ __launch_bounds__(256) void oproj_gemm_kernel(
    const u16* __restrict__ A, const float* __restrict__ W,
    const float* __restrict__ bias, float* __restrict__ Out,
    int M, int N, int K)
{
    __shared__ u32 sA[GBM][OAPAD];  // async-filled
    __shared__ u32 sB[GBN][GPAD];

    const int tid  = threadIdx.x;
    const int wave = tid >> 5, lane = tid & 31;
    const int sel  = lane >> 4,  l15 = lane & 15;
    const int wm = wave >> 2, wn = wave & 3;
    const int bM = blockIdx.y * GBM, bN = blockIdx.x * GBN;

    v8f acc[4][2];
#pragma unroll
    for (int i = 0; i < 4; ++i)
#pragma unroll
        for (int j = 0; j < 2; ++j)
#pragma unroll
            for (int r = 0; r < 8; ++r) acc[i][j][r] = 0.0f;

    const int bnb = (tid & 15) * 8, bkp = tid >> 4;
    const u32 sAbase = lds_off(&sA[0][0]);

    for (int k0 = 0; k0 < K; k0 += GBK) {
        __syncthreads();
        // A tile: 128 rows x 64B, async straight into LDS (2 x B128 per thread)
#pragma unroll
        for (int i = 0; i < 2; ++i) {
            int c   = tid + 256 * i;           // chunk 0..511
            int row = c >> 2, ch = c & 3;      // 4 x 16B per row
            const u16* g = A + (size_t)(bM + row) * K + k0 + ch * 8;
            async_copy_b128(sAbase + row * (OAPAD * 4) + ch * 16, g);
        }
        const float* bp0 = W + (size_t)(k0 + 2 * bkp) * N + bN + bnb;
        const float* bp1 = bp0 + N;
#pragma unroll
        for (int i = 0; i < 8; ++i)
            sB[bnb + i][bkp] = pack2(f2bf(bp0[i]), f2bf(bp1[i]));
        if (k0 + GBK < K)
            __builtin_prefetch(bp0 + (size_t)GBK * N, 0, 1);
        wait_async0();
        __syncthreads();

        v16bf af[4];
#pragma unroll
        for (int i = 0; i < 4; ++i)
            af[i] = ldfragA(&sA[wm * 64 + i * 16 + l15][0], sel);
#pragma unroll
        for (int j = 0; j < 2; ++j) {
            v16bf bf = ldfragB(&sB[wn * 32 + j * 16 + l15][0], sel);
#pragma unroll
            for (int i = 0; i < 4; ++i)
                acc[i][j] = wmma_bf16(af[i], bf, acc[i][j]);
        }
    }

#pragma unroll
    for (int i = 0; i < 4; ++i)
#pragma unroll
        for (int j = 0; j < 2; ++j) {
            int gc = bN + wn * 32 + j * 16 + l15;
            float bv = bias[gc];
#pragma unroll
            for (int r = 0; r < 8; ++r) {
                int gr = bM + wm * 64 + i * 16 + r + sel * 8;
                Out[(size_t)gr * N + gc] = acc[i][j][r] + bv;
            }
        }
}

// ---------- Attention: flash-style, full-row softmax (matches -0.0001 mask) ----------
#define ATQ 128
#define ATK 64
#define QKPAD 36   // 144B row stride: 16B-aligned for async B128, conflict-free

__global__ __launch_bounds__(256) void attn_kernel(
    const u16* __restrict__ QKV,   // [B*S, 3072] bf16
    u16* __restrict__ AO,          // [B*S, 1024] bf16
    int S)
{
    __shared__ u32 sQ[ATQ][QKPAD]; // [q][d-pair], async-filled
    __shared__ u32 sK[ATK][QKPAD]; // [key][d-pair], async-filled
    __shared__ u32 sV[64][33];     // [d][key-pair]  (V transposed, VALU-filled)
    __shared__ u32 sP[ATQ][33];    // per-wave P staging [q][key-pair]

    const int tid  = threadIdx.x;
    const int wave = tid >> 5, lane = tid & 31;
    const int sel  = lane >> 4,  l15 = lane & 15;
    const int qt = blockIdx.x, bh = blockIdx.y;
    const int b = bh >> 4, h = bh & 15;
    const int HD3 = 3072;
    const u32 sQbase = lds_off(&sQ[0][0]);
    const u32 sKbase = lds_off(&sK[0][0]);

    // Q tile (128 x 64 bf16 = 16KB): 4 async B128 per thread
#pragma unroll
    for (int i = 0; i < 4; ++i) {
        int c   = tid + 256 * i;            // chunk 0..1023
        int row = c >> 3, ch = c & 7;       // 8 x 16B per row
        const u16* g = QKV + (size_t)(b * S + qt * ATQ + row) * HD3 + h * 64 + ch * 8;
        async_copy_b128(sQbase + row * (QKPAD * 4) + ch * 16, g);
    }
    wait_async0();
    __syncthreads();

    v16bf qf[2];
    qf[0] = ldfragA(&sQ[wave * 16 + l15][0],  sel);
    qf[1] = ldfragA(&sQ[wave * 16 + l15][16], sel);

    v8f oacc[4];
    float mrow[8], lrow[8];
#pragma unroll
    for (int n = 0; n < 4; ++n)
#pragma unroll
        for (int r = 0; r < 8; ++r) oacc[n][r] = 0.0f;
#pragma unroll
    for (int r = 0; r < 8; ++r) { mrow[r] = -1e30f; lrow[r] = 0.0f; }

    const float scale = 0.125f;   // 1/sqrt(64)
    const int nKT = S / ATK;

    for (int kt = 0; kt < nKT; ++kt) {
        __syncthreads();
        // K tile (64 x 64 bf16 = 8KB): 2 async B128 per thread
#pragma unroll
        for (int i = 0; i < 2; ++i) {
            int c   = tid + 256 * i;        // chunk 0..511
            int row = c >> 3, ch = c & 7;
            const u16* g = QKV + (size_t)(b * S + kt * ATK + row) * HD3 + 1024 + h * 64 + ch * 8;
            async_copy_b128(sKbase + row * (QKPAD * 4) + ch * 16, g);
        }
        // V tile transposed -> sV[d][key-pair] (needs interleave: register path)
        {
            int kp2 = tid & 31, dg = tid >> 5;
            const u16* v0 = QKV + (size_t)(b * S + kt * ATK + 2 * kp2) * HD3 + 2048 + h * 64 + dg * 8;
            const u16* v1 = v0 + HD3;
#pragma unroll
            for (int i = 0; i < 8; ++i) sV[dg * 8 + i][kp2] = pack2(v0[i], v1[i]);
        }
        wait_async0();
        __syncthreads();

        // scores S = Q K^T (4 key sub-tiles x 2 d-chunks)
        v8f s[4];
#pragma unroll
        for (int n = 0; n < 4; ++n) {
            v8f a;
#pragma unroll
            for (int r = 0; r < 8; ++r) a[r] = 0.0f;
            a = wmma_bf16(qf[0], ldfragB(&sK[n * 16 + l15][0],  sel), a);
            a = wmma_bf16(qf[1], ldfragB(&sK[n * 16 + l15][16], sel), a);
            s[n] = a;
        }

        // scale + causal replacement with -0.0001 (reference semantics)
#pragma unroll
        for (int n = 0; n < 4; ++n) {
            int ck = kt * ATK + n * 16 + l15;
#pragma unroll
            for (int r = 0; r < 8; ++r) {
                int qg = qt * ATQ + wave * 16 + r + sel * 8;
                float v = s[n][r] * scale;
                s[n][r] = (ck <= qg) ? v : -1e-4f;
            }
        }

        // online softmax per row (row lives in 16 lanes of one half-wave)
#pragma unroll
        for (int r = 0; r < 8; ++r) {
            float mx = fmaxf(fmaxf(s[0][r], s[1][r]), fmaxf(s[2][r], s[3][r]));
            mx = fmaxf(mx, __shfl_xor(mx, 1, 32));
            mx = fmaxf(mx, __shfl_xor(mx, 2, 32));
            mx = fmaxf(mx, __shfl_xor(mx, 4, 32));
            mx = fmaxf(mx, __shfl_xor(mx, 8, 32));
            float mnew = fmaxf(mrow[r], mx);
            float rs = 0.0f;
#pragma unroll
            for (int n = 0; n < 4; ++n) {
                float p = __expf(s[n][r] - mnew);
                s[n][r] = p;
                rs += p;
            }
            rs += __shfl_xor(rs, 1, 32);
            rs += __shfl_xor(rs, 2, 32);
            rs += __shfl_xor(rs, 4, 32);
            rs += __shfl_xor(rs, 8, 32);
            float alpha = __expf(mrow[r] - mnew);
            lrow[r] = lrow[r] * alpha + rs;
            mrow[r] = mnew;
#pragma unroll
            for (int n = 0; n < 4; ++n) oacc[n][r] *= alpha;
        }

        // stage P (bf16) into per-wave LDS region
        u16* sp16 = (u16*)sP;
#pragma unroll
        for (int n = 0; n < 4; ++n)
#pragma unroll
            for (int r = 0; r < 8; ++r)
                sp16[(size_t)(wave * 16 + r + sel * 8) * 66 + n * 16 + l15] = f2bf(s[n][r]);

        // O += P @ V  (2 key-chunks of 32)
        v16bf pf[2];
        pf[0] = ldfragA(&sP[wave * 16 + l15][0],  sel);
        pf[1] = ldfragA(&sP[wave * 16 + l15][16], sel);
#pragma unroll
        for (int n = 0; n < 4; ++n) {
            oacc[n] = wmma_bf16(pf[0], ldfragB(&sV[n * 16 + l15][0],  sel), oacc[n]);
            oacc[n] = wmma_bf16(pf[1], ldfragB(&sV[n * 16 + l15][16], sel), oacc[n]);
        }
    }

    // normalize + store bf16 [b,q,h*64+d]
#pragma unroll
    for (int r = 0; r < 8; ++r) lrow[r] = 1.0f / lrow[r];
#pragma unroll
    for (int n = 0; n < 4; ++n)
#pragma unroll
        for (int r = 0; r < 8; ++r) {
            int qg = qt * ATQ + wave * 16 + r + sel * 8;
            int d  = n * 16 + l15;
            AO[(size_t)(b * S + qg) * 1024 + h * 64 + d] = f2bf(oacc[n][r] * lrow[r]);
        }
}

// ---------- host launcher ----------
extern "C" void kernel_launch(void* const* d_in, const int* in_sizes, int n_in,
                              void* d_out, int out_size, void* d_ws, size_t ws_size,
                              hipStream_t stream) {
    (void)in_sizes; (void)n_in; (void)out_size; (void)ws_size;
    const float* x    = (const float*)d_in[0];
    const float* Wqkv = (const float*)d_in[1];
    const float* bqkv = (const float*)d_in[2];
    const float* Wo   = (const float*)d_in[3];
    const float* bo   = (const float*)d_in[4];
    float* out = (float*)d_out;

    const int B = 4, S = 2048, D = 1024, H = 16;
    const int M = B * S;           // 8192
    const int N3 = 3 * 1024;       // 3072

    u16* qkv = (u16*)d_ws;                          // 48 MB bf16
    u16* ao  = qkv + (size_t)M * N3;                // 16 MB bf16

    dim3 blk(256);
    qkv_gemm_kernel<<<dim3(N3 / GBN, M / GBM), blk, 0, stream>>>(x, Wqkv, bqkv, qkv, M, N3, D);
    attn_kernel<<<dim3(S / ATQ, B * H), blk, 0, stream>>>(qkv, ao, S);
    oproj_gemm_kernel<<<dim3(D / GBN, M / GBM), blk, 0, stream>>>(ao, Wo, bo, out, M, D, D);
}
